// GATRec_8641474200165
// MI455X (gfx1250) — matrix operand order repaired and checked by reading, not compiled, not run on hardware
//
#include <hip/hip_runtime.h>
#include <hip/hip_bf16.h>

#define HID 256
#define SEQ 256
#define BAT 256
#define ALPHA 0.2f

typedef __bf16 v16bf __attribute__((ext_vector_type(16)));
typedef float  v8f   __attribute__((ext_vector_type(8)));

union Frag {
    v16bf v;
    unsigned int u[8];
};

// ---------------------------------------------------------------------------
// k0: W [in=h][out=k] f32  ->  WT [k][h] bf16  (so B-frag K-pairs are adjacent)
// ---------------------------------------------------------------------------
__global__ __launch_bounds__(256) void k0_wt(const float* __restrict__ Wf,
                                             __bf16* __restrict__ WT) {
    int k = blockIdx.x;
    int h = threadIdx.x;
    WT[k * HID + h] = (__bf16)Wf[h * HID + k];
}

// ---------------------------------------------------------------------------
// k1: gather + layernorm + projection  h = LN(x) @ W
//     writes hT[b][feat][row] bf16 (transposed for stage-2 B fragments)
// block = 128 threads (4 waves), grid = (SEQ/16, BAT)
// ---------------------------------------------------------------------------
__global__ __launch_bounds__(128) void k1_proj(
    const int*   __restrict__ seq,
    const float* __restrict__ emb,
    const float* __restrict__ pos,
    const __bf16* __restrict__ WT,
    const float* __restrict__ gamma,
    const float* __restrict__ beta,
    __bf16* __restrict__ hT) {

    __shared__ __align__(16) __bf16 xs[16 * HID];   // normalized rows, 8KB

    const int b       = blockIdx.y;
    const int rowBase = blockIdx.x * 16;
    const int lane    = threadIdx.x & 31;
    const int wave    = threadIdx.x >> 5;
    const int half    = lane >> 4;
    const int l16     = lane & 15;

    // ---- Phase A: gather + pos + layernorm, 4 rows per wave ----
    for (int t = 0; t < 4; ++t) {
        const int r   = wave * 4 + t;            // local row 0..15
        const int l   = rowBase + r;             // seq position
        const int idx = seq[b * SEQ + l];
        float v[8];
        float s = 0.f, ss = 0.f;
#pragma unroll
        for (int k = 0; k < 8; ++k) {
            const int e = k * 32 + lane;
            const float x = emb[(size_t)idx * HID + e] + pos[l * HID + e];
            v[k] = x; s += x; ss += x * x;
        }
#pragma unroll
        for (int m = 16; m >= 1; m >>= 1) {
            s  += __shfl_xor(s,  m, 32);
            ss += __shfl_xor(ss, m, 32);
        }
        const float mu  = s * (1.f / HID);
        const float var = ss * (1.f / HID) - mu * mu;
        const float rsd = rsqrtf(var + 1e-12f);
#pragma unroll
        for (int k = 0; k < 8; ++k) {
            const int e = k * 32 + lane;
            xs[r * HID + e] = (__bf16)((v[k] - mu) * rsd * gamma[e] + beta[e]);
        }
    }
    __syncthreads();

    // ---- Phase B: 16x256 tile @ 256x256 via v_wmma_f32_16x16x32_bf16 ----
    const unsigned int* xsw = (const unsigned int*)xs;
    v8f acc[4] = {{0,0,0,0,0,0,0,0}, {0,0,0,0,0,0,0,0},
                  {0,0,0,0,0,0,0,0}, {0,0,0,0,0,0,0,0}};

    const int akb = half ? 16 : 0;   // A: lanes 16-31 carry K=16..31
    const int bkb = half ? 16 : 0;   // B: lanes 16-31 carry K=16..31

    for (int k0 = 0; k0 < HID; k0 += 32) {
        Frag A;
#pragma unroll
        for (int j = 0; j < 4; ++j)
            A.u[j]     = xsw[(l16 * HID + k0 + akb + 2 * j) >> 1];
#pragma unroll
        for (int j = 0; j < 4; ++j)
            A.u[4 + j] = xsw[(l16 * HID + k0 + akb + 8 + 2 * j) >> 1];

#pragma unroll
        for (int nt = 0; nt < 4; ++nt) {
            const int n = (nt * 4 + wave) * 16 + l16;       // output feature
            const unsigned int* wrow = (const unsigned int*)(WT + (size_t)n * HID);
            Frag Bm;
#pragma unroll
            for (int j = 0; j < 8; ++j)
                Bm.u[j] = wrow[(k0 + bkb + 2 * j) >> 1];
            acc[nt] = __builtin_amdgcn_wmma_f32_16x16x32_bf16(
                false, A.v, false, Bm.v, (short)0, acc[nt], false, false);
        }
    }

    // ---- store hT[b][n][row] (D layout: M = r + 8*half, N = l16) ----
#pragma unroll
    for (int nt = 0; nt < 4; ++nt) {
        const int n = (nt * 4 + wave) * 16 + l16;
#pragma unroll
        for (int r = 0; r < 8; ++r) {
            const int row = rowBase + r + half * 8;
            hT[((size_t)b * HID + n) * SEQ + row] = (__bf16)acc[nt][r];
        }
    }
}

// ---------------------------------------------------------------------------
// k2: s1 = h . a1 ; s2 = h . a2   (coalesced over hT's row dimension)
// ---------------------------------------------------------------------------
__global__ __launch_bounds__(256) void k2_scores(
    const __bf16* __restrict__ hT,
    const float*  __restrict__ a1,
    const float*  __restrict__ a2,
    float* __restrict__ s1, float* __restrict__ s2) {
    const int b = blockIdx.x;
    const int j = threadIdx.x;
    const __bf16* base = hT + (size_t)b * HID * SEQ;
    float r1 = 0.f, r2 = 0.f;
    for (int f = 0; f < HID; ++f) {
        const float h = (float)base[f * SEQ + j];
        r1 += h * a1[f];
        r2 += h * a2[f];
    }
    s1[b * SEQ + j] = r1;
    s2[b * SEQ + j] = r2;
}

// ---------------------------------------------------------------------------
// k3: softmax(leaky(s1_i + s2_j)) @ h, normalized post-GEMM by row-sum
// block = 128 threads (4 waves), grid = (SEQ/16, BAT)
// ---------------------------------------------------------------------------
__global__ __launch_bounds__(128) void k3_attn(
    const __bf16* __restrict__ hT,
    const float*  __restrict__ s1g,
    const float*  __restrict__ s2g,
    float* __restrict__ out) {

    __shared__ __align__(16) float  s2s[SEQ];       // 1KB
    __shared__ __align__(16) float  rs[16];
    __shared__ __align__(16) __bf16 attn[16 * SEQ]; // unnormalized exp, 8KB

    const int b       = blockIdx.y;
    const int rowBase = blockIdx.x * 16;
    const int lane    = threadIdx.x & 31;
    const int wave    = threadIdx.x >> 5;
    const int half    = lane >> 4;
    const int l16     = lane & 15;

    for (int t = threadIdx.x; t < SEQ; t += 128) s2s[t] = s2g[b * SEQ + t];
    __syncthreads();

    // global max of s2 (per-wave redundant reduce) -> analytic row max
    float mx = -1e30f;
#pragma unroll
    for (int k = 0; k < 8; ++k) mx = fmaxf(mx, s2s[k * 32 + lane]);
#pragma unroll
    for (int m = 16; m >= 1; m >>= 1) mx = fmaxf(mx, __shfl_xor(mx, m, 32));

    for (int t = 0; t < 4; ++t) {
        const int r  = wave * 4 + t;
        const int i  = rowBase + r;
        const float si = s1g[b * SEQ + i];
        const float zm = si + mx;
        const float m0 = zm > 0.f ? zm : ALPHA * zm;   // leaky monotone -> true row max
        float sum = 0.f;
#pragma unroll
        for (int k = 0; k < 8; ++k) {
            const int j = k * 32 + lane;
            const float z = si + s2s[j];
            const float e = z > 0.f ? z : ALPHA * z;
            const float p = __expf(e - m0);
            sum += p;
            attn[r * SEQ + j] = (__bf16)p;
        }
#pragma unroll
        for (int m = 16; m >= 1; m >>= 1) sum += __shfl_xor(sum, m, 32);
        if (lane == 0) rs[r] = sum;
    }
    __syncthreads();

    // ---- GEMM2: 16xSEQ attn (LDS, bf16) @ SEQxHID h (hT global, bf16) ----
    const unsigned int* aw = (const unsigned int*)attn;
    v8f acc[4] = {{0,0,0,0,0,0,0,0}, {0,0,0,0,0,0,0,0},
                  {0,0,0,0,0,0,0,0}, {0,0,0,0,0,0,0,0}};
    const int akb = half ? 16 : 0;
    const int bkb = half ? 16 : 0;

    for (int k0 = 0; k0 < SEQ; k0 += 32) {
        Frag A;
#pragma unroll
        for (int j = 0; j < 4; ++j)
            A.u[j]     = aw[(l16 * SEQ + k0 + akb + 2 * j) >> 1];
#pragma unroll
        for (int j = 0; j < 4; ++j)
            A.u[4 + j] = aw[(l16 * SEQ + k0 + akb + 8 + 2 * j) >> 1];

#pragma unroll
        for (int nt = 0; nt < 4; ++nt) {
            const int n = (nt * 4 + wave) * 16 + l16;   // output feature
            const unsigned int* hrow =
                (const unsigned int*)(hT + ((size_t)b * HID + n) * SEQ);
            Frag Bm;
#pragma unroll
            for (int j = 0; j < 8; ++j)
                Bm.u[j] = hrow[(k0 + bkb + 2 * j) >> 1];
            acc[nt] = __builtin_amdgcn_wmma_f32_16x16x32_bf16(
                false, A.v, false, Bm.v, (short)0, acc[nt], false, false);
        }
    }

    // normalize by row-sum and store f32 output [B][L][H]
#pragma unroll
    for (int nt = 0; nt < 4; ++nt) {
        const int n = (nt * 4 + wave) * 16 + l16;
#pragma unroll
        for (int r = 0; r < 8; ++r) {
            const int rl  = r + half * 8;                 // local row 0..15
            const int row = rowBase + rl;
            out[((size_t)b * SEQ + row) * HID + n] = acc[nt][r] / rs[rl];
        }
    }
}

// ---------------------------------------------------------------------------
extern "C" void kernel_launch(void* const* d_in, const int* in_sizes, int n_in,
                              void* d_out, int out_size, void* d_ws, size_t ws_size,
                              hipStream_t stream) {
    const int*   seq   = (const int*)  d_in[0];
    const float* emb   = (const float*)d_in[1];
    const float* pos   = (const float*)d_in[2];
    const float* Wf    = (const float*)d_in[3];
    const float* a1    = (const float*)d_in[4];
    const float* a2    = (const float*)d_in[5];
    const float* gamma = (const float*)d_in[6];
    const float* beta  = (const float*)d_in[7];
    float* out = (float*)d_out;

    // workspace layout
    char* ws = (char*)d_ws;
    __bf16* WT = (__bf16*)ws;                                  // 128 KB
    __bf16* hT = (__bf16*)(ws + (size_t)131072);               // 32 MB
    float*  s1 = (float*)(ws + (size_t)131072 + 33554432);     // 256 KB
    float*  s2 = s1 + BAT * SEQ;                               // 256 KB

    k0_wt<<<dim3(HID), dim3(HID), 0, stream>>>(Wf, WT);
    k1_proj<<<dim3(SEQ / 16, BAT), dim3(128), 0, stream>>>(
        seq, emb, pos, WT, gamma, beta, hT);
    k2_scores<<<dim3(BAT), dim3(SEQ), 0, stream>>>(hT, a1, a2, s1, s2);
    k3_attn<<<dim3(SEQ / 16, BAT), dim3(128), 0, stream>>>(hT, s1, s2, out);
}
